// MoEAdapter_82437602279462
// MI455X (gfx1250) — matrix operand order repaired and checked by reading, not compiled, not run on hardware
//
#include <hip/hip_runtime.h>
#include <hip/hip_bf16.h>

typedef __bf16 bf16_t;
typedef __attribute__((ext_vector_type(16))) __bf16 v16bf;
typedef __attribute__((ext_vector_type(8)))  __bf16 v8bf;
typedef __attribute__((ext_vector_type(8)))  float   v8f;
typedef __attribute__((ext_vector_type(4)))  float   v4f;

#define M_TOTAL 32768
#define D_DIM   1024
#define E_NUM   8
#define R_DIM   16
#define EC      128          // E*R
#define M_TILE  64
#define KC      64           // K chunk staged in LDS (f32, double buffered)
#define NCHUNK  (D_DIM / KC) // 16
#define KSTEPS  (KC / 32)    // 2

// workspace offsets in ushort (bf16) units; all multiples of 512
#define WDS_OFF 0            // [K=1024,N=128]  32x8 blocks  -> 131072
#define WUS_OFF 131072       // [K=128, N=1024]  4x64 blocks -> 131072
#define WRS_OFF 262144       // [K=1024,N=16]   32x1 blocks  -> 16384
#define BUS_OFF 278528       // [K=32,  N=1024]  1x64 blocks -> 32768
#define WS_TOTAL 311296

#if __has_builtin(__builtin_amdgcn_sched_group_barrier)
#define SGB(mask, size, id) __builtin_amdgcn_sched_group_barrier(mask, size, id)
#else
#define SGB(mask, size, id)
#endif
// masks: 0x002 VALU, 0x008 MFMA/WMMA, 0x020 VMEM read, 0x040 VMEM write, 0x100 DS read

template<bool B> struct BoolC { static constexpr bool value = B; };

static __device__ __forceinline__ unsigned short f2b(float x) {
    bf16_t b = (bf16_t)x;
    return __builtin_bit_cast(unsigned short, b);
}

// Build bf16 weights in pre-swizzled B-fragment block layout:
// block = 32(K) x 16(N); within a block, lane l owns 16 contiguous halfs:
//   K = kt*32 + (h>=8 ? 16 : 0) + (l>=16 ? 8 : 0) + (h & 7),  N = nt*16 + (l & 15)
// so a wave loads one whole B fragment with one 32-byte read per lane.
__global__ __launch_bounds__(256) void moe_prep(
    const float* __restrict__ Wr, const float* __restrict__ Wd,
    const float* __restrict__ Wu, const float* __restrict__ bu,
    unsigned short* __restrict__ ws)
{
    int t = blockIdx.x * 256 + threadIdx.x;
    if (t >= WS_TOTAL) return;
    int r    = t & 511;
    int lane = r >> 4;
    int h    = r & 15;
    int kin  = ((h >> 3) << 4) + ((lane >> 4) << 3) + (h & 7);  // K offset within 32-row block
    int nin  = lane & 15;

    if (t < 131072) {                       // WdS: Wd (E,D,R) -> [D, E*R], KT=32
        int bi = t >> 9;
        int nt = bi >> 5, kt = bi & 31;
        int K = kt * 32 + kin, N = nt * 16 + nin;
        float v = Wd[(size_t)(N >> 4) * (D_DIM * R_DIM) + (size_t)K * R_DIM + (N & 15)];
        ws[WDS_OFF + t] = f2b(v);
    } else if (t < 262144) {                // WuS: Wu (E,R,D) == [128,1024], KT=4
        int t2 = t - 131072;
        int bi = t2 >> 9;
        int nt = bi >> 2, kt = bi & 3;
        int K = kt * 32 + kin, N = nt * 16 + nin;
        ws[WUS_OFF + t2] = f2b(Wu[(size_t)K * D_DIM + N]);
    } else if (t < 278528) {                // WrS: [1024,16] (cols>=8 zero), KT=32
        int t2 = t - 262144;
        int kt = t2 >> 9;
        int K = kt * 32 + kin, N = nin;
        float v = (N < E_NUM) ? Wr[(size_t)K * E_NUM + N] : 0.f;
        ws[WRS_OFF + t2] = f2b(v);
    } else {                                // BuS: [32,1024] (rows>=8 zero), KT=1
        int t2 = t - 278528;
        int nt = t2 >> 9;
        int K = kin, N = nt * 16 + nin;
        float v = (K < E_NUM) ? bu[(size_t)K * D_DIM + N] : 0.f;
        ws[BUS_OFF + t2] = f2b(v);
    }
}

// ---- CDNA5 async copy: global -> LDS, 16B per lane, ASYNCcnt-tracked ----
static __device__ __forceinline__ void async_cp_b128(void* lds_ptr, const void* gptr) {
    unsigned lds_off = (unsigned)(uintptr_t)lds_ptr;   // flat->LDS: low 32 bits
    unsigned long long ga = (unsigned long long)(uintptr_t)gptr;
    asm volatile("global_load_async_to_lds_b128 %0, %1, off"
                 :: "v"(lds_off), "v"(ga) : "memory");
}
static __device__ __forceinline__ void wait_async0() {
    asm volatile("s_wait_asynccnt 0x0" ::: "memory");
}

// A fragment (16-bit, 16x32) from an f32 row in LDS; caller pre-offsets p by
// (lane>=16 ? 8 : 0). Lane takes two 8-element K runs: [0..7] and [16..23].
static __device__ __forceinline__ v16bf load_a_frag_f32(const float* p) {
    v4f a = *(const v4f*)p;
    v4f b = *(const v4f*)(p + 4);
    v4f c = *(const v4f*)(p + 16);
    v4f d = *(const v4f*)(p + 20);
    v16bf r;
    r[0]=(bf16_t)a.x; r[1]=(bf16_t)a.y; r[2]=(bf16_t)a.z; r[3]=(bf16_t)a.w;
    r[4]=(bf16_t)b.x; r[5]=(bf16_t)b.y; r[6]=(bf16_t)b.z; r[7]=(bf16_t)b.w;
    r[8]=(bf16_t)c.x; r[9]=(bf16_t)c.y; r[10]=(bf16_t)c.z; r[11]=(bf16_t)c.w;
    r[12]=(bf16_t)d.x; r[13]=(bf16_t)d.y; r[14]=(bf16_t)d.z; r[15]=(bf16_t)d.w;
    return r;
}

// A fragment from a bf16 row in LDS (caller pre-offsets by lane K base).
static __device__ __forceinline__ v16bf load_a_frag_bf16(const bf16_t* p) {
    v8bf lo = *(const v8bf*)p;
    v8bf hi = *(const v8bf*)(p + 16);
    return __builtin_shufflevector(lo, hi, 0,1,2,3,4,5,6,7,8,9,10,11,12,13,14,15);
}

static __device__ __forceinline__ v8f wmma_bf16(v16bf a, v16bf b, v8f c) {
    return __builtin_amdgcn_wmma_f32_16x16x32_bf16(false, a, false, b, (short)0, c,
                                                   false, false);
}

static __device__ __forceinline__ float gelu_tanh(float t) {
    // tanh(u) = 1 - 2/(e^{2u}+1): overflow-safe, one hardware transcendental
    float u  = 0.7978845608028654f * (t + 0.044715f * t * t * t);
    float th = 1.f - 2.f / (__expf(2.f * u) + 1.f);
    return 0.5f * t * (1.f + th);
}

__global__ __launch_bounds__(256) void moe_main(
    const float* __restrict__ x, const float* __restrict__ br,
    const float* __restrict__ bd, const unsigned short* __restrict__ ws,
    float* __restrict__ out)
{
    __shared__ float  xsf[2][M_TILE * KC];   // 32 KB  x chunks, f32, dbl-buffered
    __shared__ bf16_t hs[M_TILE * EC];       // 16 KB  gated gelu activations
    __shared__ bf16_t wpad[M_TILE * 32];     //  4 KB  gate weights, K-padded
    __shared__ float  lg[M_TILE * 16];       //  4 KB  router logits
    __shared__ float  wls[M_TILE * E_NUM];   //  2 KB  gate weights f32
    __shared__ float  bdl[EC];               // 512 B  down-proj bias

    const int tid    = threadIdx.x;
    const int lane   = tid & 31;
    const int w      = tid >> 5;
    const int mstrip = w & 3;                 // 4 row-strips of 16 tokens
    const int ntb    = (w >> 2) << 2;         // down-proj N-tile base: 0 or 4
    const int m0     = blockIdx.x * M_TILE;
    const int arow   = lane & 15;             // fragment row / col-in-tile
    const int abase  = (lane >> 4) << 3;      // 0 or 8 (A K-offset & C M-offset)
    const bool is_router = __builtin_amdgcn_readfirstlane(w) < 4;  // scalar branch

    if (tid >= 64 && tid < 64 + EC) bdl[tid - 64] = bd[tid - 64];

    // per-thread async slice: 4 x 16B per chunk (64 rows x 64 f32 = 16 KB)
    const int xrow = (tid >> 4);                    // rows 0..15 (x16 per i-step)
    const int xc4  = (tid & 15) << 2;               // float4 column
    const float* xg = x + (size_t)(m0 + xrow) * D_DIM + xc4;

    v8f rc = {};                              // router accumulator (waves 0-3)
    v8f dc[4] = {{}, {}, {}, {}};             // down-proj accumulators

    // prime the pipeline: chunk 0 -> buffer 0
    #pragma unroll
    for (int i = 0; i < 4; ++i)
        async_cp_b128(&xsf[0][(xrow + i * 16) * KC + xc4],
                      xg + (size_t)i * 16 * D_DIM);

    // ---- Phase 1: stream x (async DMA), router + all-expert down proj ----
    // warp-specialized clone: router flag is constexpr inside, so router waves
    // get a 10-load/5-WMMA body and the rest an 8-load/4-WMMA body, no per-
    // iteration predicate.  Both clones hit the same barrier count.
    auto phase1 = [&](auto RC) {
        constexpr bool ROUTER = decltype(RC)::value;
        for (int kc = 0; kc < NCHUNK; ++kc) {
            wait_async0();          // our chunk-kc copies landed
            __syncthreads();        // everyone's landed; prev compute done
            if (kc + 1 < NCHUNK) {  // prefetch next chunk into the other buffer
                const float* xn = xg + (kc + 1) * KC;
                float* dst = &xsf[(kc + 1) & 1][0];
                #pragma unroll
                for (int i = 0; i < 4; ++i)
                    async_cp_b128(dst + (xrow + i * 16) * KC + xc4,
                                  xn + (size_t)i * 16 * D_DIM);
            }
            const float* xrowp = &xsf[kc & 1][(mstrip * 16 + arow) * KC];
            #pragma unroll
            for (int ks = 0; ks < KSTEPS; ++ks) {
                int kt = kc * KSTEPS + ks;    // global 32-K block index
                v16bf b0 = *(const v16bf*)(ws + WDS_OFF + (((ntb+0) * 32 + kt) << 9) + (lane << 4));
                v16bf b1 = *(const v16bf*)(ws + WDS_OFF + (((ntb+1) * 32 + kt) << 9) + (lane << 4));
                v16bf b2 = *(const v16bf*)(ws + WDS_OFF + (((ntb+2) * 32 + kt) << 9) + (lane << 4));
                v16bf b3 = *(const v16bf*)(ws + WDS_OFF + (((ntb+3) * 32 + kt) << 9) + (lane << 4));
                v16bf b4;
                if constexpr (ROUTER)
                    b4 = *(const v16bf*)(ws + WRS_OFF + (kt << 9) + (lane << 4));
                v16bf a  = load_a_frag_f32(xrowp + ks * 32 + abase);
                dc[0] = wmma_bf16(a, b0, dc[0]);
                dc[1] = wmma_bf16(a, b1, dc[1]);
                dc[2] = wmma_bf16(a, b2, dc[2]);
                dc[3] = wmma_bf16(a, b3, dc[3]);
                if constexpr (ROUTER) rc = wmma_bf16(a, b4, rc);
                // pin shape: loads clustered, then cvts, then WMMA chain
                SGB(0x020, ROUTER ? 10 : 8, 0);   // VMEM reads (B frags)
                SGB(0x100, 4, 0);                 // DS reads (A frag)
                SGB(0x002, 8, 0);                 // cvt_pk VALUs
                SGB(0x008, ROUTER ? 5 : 4, 0);    // WMMA chain
            }
        }
    };
    if (is_router) phase1(BoolC<true>{});
    else           phase1(BoolC<false>{});

    if (is_router) {                          // dump router logits (C layout)
        #pragma unroll
        for (int v = 0; v < 8; ++v)
            lg[(mstrip * 16 + v + abase) * 16 + arow] = rc[v];
    }
    __syncthreads();

    // ---- Softmax + top-2 renorm == softmax over the 2 largest logits ----
    if (tid < M_TILE) {
        float l[E_NUM];
        #pragma unroll
        for (int e = 0; e < E_NUM; ++e) l[e] = lg[tid * 16 + e] + br[e];
        int i1 = 0; float m1 = l[0];
        #pragma unroll
        for (int e = 1; e < E_NUM; ++e) if (l[e] > m1) { m1 = l[e]; i1 = e; }
        int i2 = -1; float m2 = -3.4e38f;
        #pragma unroll
        for (int e = 0; e < E_NUM; ++e) if (e != i1 && l[e] > m2) { m2 = l[e]; i2 = e; }
        float w1 = 1.f / (1.f + __expf(m2 - m1));
        float w2 = 1.f - w1;
        #pragma unroll
        for (int e = 0; e < E_NUM; ++e) {
            float we = (e == i1) ? w1 : ((e == i2) ? w2 : 0.f);
            wls[tid * E_NUM + e] = we;
            wpad[tid * 32 + e]   = (bf16_t)we;
        }
        #pragma unroll
        for (int e = E_NUM; e < 32; ++e) wpad[tid * 32 + e] = (bf16_t)0.f;
    }
    __syncthreads();

    // ---- bias + gelu + gate, store bf16 activations ----
    #pragma unroll
    for (int i = 0; i < 4; ++i) {
        int col = (ntb + i) * 16 + arow;
        float bdv = bdl[col];
        #pragma unroll
        for (int v = 0; v < 8; ++v) {
            int row = mstrip * 16 + v + abase;
            float g = gelu_tanh(dc[i][v] + bdv);
            hs[row * EC + col] = (bf16_t)(g * wls[row * E_NUM + (col >> 4)]);
        }
    }
    __syncthreads();

    // ---- Phase 2: out = x + Hw @ Wu + w @ bu (software-pipelined) ----
    v16bf ha[4], wa;
    #pragma unroll
    for (int j = 0; j < 4; ++j)
        ha[j] = load_a_frag_bf16(&hs[(mstrip * 16 + arow) * EC + j * 32 + abase]);
    wa = load_a_frag_bf16(&wpad[(mstrip * 16 + arow) * 32 + abase]);

    const int ntoff = w >> 2;                 // interleave two waves per strip
    // prologue: load tile 0's B fragments
    v16bf nb0, nb1, nb2, nb3, nb4;
    {
        int nt = ntoff;
        nb0 = *(const v16bf*)(ws + WUS_OFF + (((nt << 2) + 0) << 9) + (lane << 4));
        nb1 = *(const v16bf*)(ws + WUS_OFF + (((nt << 2) + 1) << 9) + (lane << 4));
        nb2 = *(const v16bf*)(ws + WUS_OFF + (((nt << 2) + 2) << 9) + (lane << 4));
        nb3 = *(const v16bf*)(ws + WUS_OFF + (((nt << 2) + 3) << 9) + (lane << 4));
        nb4 = *(const v16bf*)(ws + BUS_OFF + (nt << 9) + (lane << 4));
    }
    for (int ii = 0; ii < 32; ++ii) {
        int nt  = ntoff + (ii << 1);
        int ntn = ntoff + (((ii + 1) & 31) << 1);   // branchless wrap
        v16bf b0 = nb0, b1 = nb1, b2 = nb2, b3 = nb3, b4 = nb4;
        // prefetch next tile's B fragments while this tile's WMMAs run
        nb0 = *(const v16bf*)(ws + WUS_OFF + (((ntn << 2) + 0) << 9) + (lane << 4));
        nb1 = *(const v16bf*)(ws + WUS_OFF + (((ntn << 2) + 1) << 9) + (lane << 4));
        nb2 = *(const v16bf*)(ws + WUS_OFF + (((ntn << 2) + 2) << 9) + (lane << 4));
        nb3 = *(const v16bf*)(ws + WUS_OFF + (((ntn << 2) + 3) << 9) + (lane << 4));
        nb4 = *(const v16bf*)(ws + BUS_OFF + (ntn << 9) + (lane << 4));
        // residual x loads for this tile (independent of WMMA)
        float xr[8];
        #pragma unroll
        for (int v = 0; v < 8; ++v)
            xr[v] = x[(size_t)(m0 + mstrip * 16 + v + abase) * D_DIM + nt * 16 + arow];
        v8f c = {};
        c = wmma_bf16(ha[0], b0, c);
        c = wmma_bf16(ha[1], b1, c);
        c = wmma_bf16(ha[2], b2, c);
        c = wmma_bf16(ha[3], b3, c);
        c = wmma_bf16(wa,    b4, c);          // += w @ bu (bu K-padded to 32)
        #pragma unroll
        for (int v = 0; v < 8; ++v) {
            size_t g = (size_t)(m0 + mstrip * 16 + v + abase) * D_DIM + nt * 16 + arow;
            out[g] = xr[v] + c[v];
        }
        SGB(0x020, 18, 0);  // next-tile B (10) + residual x (8) reads
        SGB(0x008, 5, 0);   // WMMA chain
        SGB(0x002, 8, 0);   // residual adds
        SGB(0x040, 8, 0);   // stores
    }
}

extern "C" void kernel_launch(void* const* d_in, const int* in_sizes, int n_in,
                              void* d_out, int out_size, void* d_ws, size_t ws_size,
                              hipStream_t stream) {
    const float* x  = (const float*)d_in[0];
    const float* Wr = (const float*)d_in[1];
    const float* br = (const float*)d_in[2];
    const float* Wd = (const float*)d_in[3];
    const float* bd = (const float*)d_in[4];
    const float* Wu = (const float*)d_in[5];
    const float* bu = (const float*)d_in[6];
    unsigned short* ws = (unsigned short*)d_ws;

    moe_prep<<<WS_TOTAL / 256, 256, 0, stream>>>(Wr, Wd, Wu, bu, ws);
    moe_main<<<M_TOTAL / M_TILE, 256, 0, stream>>>(x, br, bd, ws, (float*)d_out);
}
